// CodecAttention_33174327394429
// MI455X (gfx1250) — matrix operand order repaired and checked by reading, not compiled, not run on hardware
//
#include <hip/hip_runtime.h>
#include <hip/hip_bf16.h>

// ---------------------------------------------------------------------------
// Problem constants (match reference)
// ---------------------------------------------------------------------------
#define Bc      4
#define Tc      2048
#define DIMc    1024
#define NHEADS  8
#define HDIM    128
#define WIN     16
#define ROWS    (Bc * Tc)          // 8192
#define NQKV    (3 * DIMc)         // 3072

typedef __attribute__((ext_vector_type(16))) __bf16 bf16x16;
typedef __attribute__((ext_vector_type(8)))  __bf16 bf16x8;
typedef __attribute__((ext_vector_type(8)))  float  f32x8;

// Async copy of 16B from global to LDS (CDNA5 GLOBAL_LOAD_ASYNC_TO_LDS_B128,
// GV addressing). ldsoff = LDS byte offset (low 32 bits of flat LDS address).
__device__ __forceinline__ void async_b128(unsigned ldsoff, const void* gaddr) {
  asm volatile("global_load_async_to_lds_b128 %0, %1, off"
               :: "v"(ldsoff), "v"((unsigned long long)(size_t)gaddr)
               : "memory");
}
__device__ __forceinline__ void wait_async(void) {
  asm volatile("s_wait_asynccnt 0x0" ::: "memory");
}
__device__ __forceinline__ void wait_async4(void) {
  asm volatile("s_wait_asynccnt 0x4" ::: "memory");
}
__device__ __forceinline__ unsigned lds_off(const void* p) {
  return (unsigned)(size_t)p;   // flat LDS address low 32 bits == LDS offset
}

// ---------------------------------------------------------------------------
// Elementwise converts
// ---------------------------------------------------------------------------
__global__ __launch_bounds__(256) void k_conv_x(
    const float* __restrict__ x, __bf16* __restrict__ xb) {
  const size_t i = ((size_t)blockIdx.x * 256 + threadIdx.x) * 4;
  const float4 v = *(const float4*)(x + i);
  xb[i + 0] = (__bf16)v.x;
  xb[i + 1] = (__bf16)v.y;
  xb[i + 2] = (__bf16)v.z;
  xb[i + 3] = (__bf16)v.w;
}

// Weight convert + transpose: Wt[n][k] = W[k][n] as bf16
__global__ __launch_bounds__(256) void k_conv_wqkv(
    const float* __restrict__ wq, const float* __restrict__ wk,
    const float* __restrict__ wv, __bf16* __restrict__ wt) {
  int idx = blockIdx.x * 256 + threadIdx.x;   // 3072*1024 total
  int n = idx >> 10;
  int k = idx & (DIMc - 1);
  const float* w = (n < DIMc) ? wq : (n < 2 * DIMc ? wk : wv);
  int nn = n & (DIMc - 1);
  wt[idx] = (__bf16)w[(size_t)k * DIMc + nn];
}

__global__ __launch_bounds__(256) void k_conv_wo(
    const float* __restrict__ wo, __bf16* __restrict__ wt) {
  int idx = blockIdx.x * 256 + threadIdx.x;   // 1024*1024 total
  int n = idx >> 10;
  int k = idx & (DIMc - 1);
  wt[idx] = (__bf16)wo[(size_t)k * DIMc + n];
}

// ---------------------------------------------------------------------------
// WMMA GEMM: C[M][N] (f32) = A[M][K] (bf16) * Bt[N][K]^T (bf16, pre-transposed)
// Block tile 128x128x32, 8 waves (4x2), wave tile 32x64 = 2x4 wmma accums.
// Double-buffered LDS filled with GLOBAL_LOAD_ASYNC_TO_LDS_B128 (ASYNCcnt).
// Fragment layouts follow CDNA5 ISA 7.12.2 (wave32).
// ---------------------------------------------------------------------------
__global__ __launch_bounds__(256) void k_gemm_wmma(
    const __bf16* __restrict__ A, const __bf16* __restrict__ Bt,
    float* __restrict__ C, int M, int N, int K) {
  constexpr int BM = 128, BN = 128, BK = 32, PAD = 8;   // PAD keeps 16B align
  __shared__ __align__(16) __bf16 As[2][BM][BK + PAD];
  __shared__ __align__(16) __bf16 Bs[2][BN][BK + PAD];

  const int m0 = blockIdx.y * BM;
  const int n0 = blockIdx.x * BN;
  const int tid = threadIdx.x;
  const int w = tid >> 5, lane = tid & 31;
  const int wm = (w >> 1) * 32;        // wave row offset in block tile
  const int wn = (w & 1) * 64;         // wave col offset in block tile
  const int l15 = lane & 15;
  const int kb_a = (lane >> 4) * 8;    // A frag K base (ISA A 16x32 layout)
  const int kb_b = (lane >> 4) * 8;    // B frag K base (same layout as A)
  const int mh = (lane >> 4) * 8;      // D frag M half offset

  // staging: thread -> row (tid>>1), 32B half (tid&1), as 2x16B async ops
  const int sr = tid >> 1;             // 0..127
  const int sc0 = (tid & 1) * 16;      // bf16 col 0 or 16

  const __bf16* aRow = A + (size_t)(m0 + sr) * K + sc0;
  const __bf16* bRow = Bt + (size_t)(n0 + sr) * K + sc0;

  f32x8 acc[2][4] = {};
  union Frag { bf16x16 v; bf16x8 h[2]; };

  const int NT = K / BK;

  // prologue: issue tile 0 into buffer 0 (4 async instrs per wave)
  async_b128(lds_off(&As[0][sr][sc0 + 0]), aRow + 0);
  async_b128(lds_off(&As[0][sr][sc0 + 8]), aRow + 8);
  async_b128(lds_off(&Bs[0][sr][sc0 + 0]), bRow + 0);
  async_b128(lds_off(&Bs[0][sr][sc0 + 8]), bRow + 8);

  for (int it = 0; it < NT; ++it) {
    const int cur = it & 1;
    if (it + 1 < NT) {
      const int nxt = cur ^ 1;
      const int kn = (it + 1) * BK;
      async_b128(lds_off(&As[nxt][sr][sc0 + 0]), aRow + kn + 0);
      async_b128(lds_off(&As[nxt][sr][sc0 + 8]), aRow + kn + 8);
      async_b128(lds_off(&Bs[nxt][sr][sc0 + 0]), bRow + kn + 0);
      async_b128(lds_off(&Bs[nxt][sr][sc0 + 8]), bRow + kn + 8);
      wait_async4();       // tile `it` complete; tile `it+1` still in flight
    } else {
      wait_async();
    }
    __syncthreads();       // tile `it` visible to all waves

    Frag a[2], bfr[4];
#pragma unroll
    for (int i = 0; i < 2; ++i) {
      const __bf16* row = &As[cur][wm + i * 16 + l15][0];
      a[i].h[0] = *(const bf16x8*)(row + kb_a);        // K = kb..kb+7
      a[i].h[1] = *(const bf16x8*)(row + kb_a + 16);   // K = kb+16..kb+23
    }
#pragma unroll
    for (int j = 0; j < 4; ++j) {
      const __bf16* row = &Bs[cur][wn + j * 16 + l15][0];
      bfr[j].h[0] = *(const bf16x8*)(row + kb_b);      // K = kb..kb+7
      bfr[j].h[1] = *(const bf16x8*)(row + kb_b + 16); // K = kb+16..kb+23
    }
#pragma unroll
    for (int i = 0; i < 2; ++i)
#pragma unroll
      for (int j = 0; j < 4; ++j)
        acc[i][j] = __builtin_amdgcn_wmma_f32_16x16x32_bf16(
            false, a[i].v, false, bfr[j].v, (short)0, acc[i][j], false, false);
    __syncthreads();       // readers done before buffer `cur` is refilled
  }

  // store D: VGPR r holds M = r + (lane>=16 ? 8 : 0), N = lane&15
#pragma unroll
  for (int i = 0; i < 2; ++i) {
#pragma unroll
    for (int j = 0; j < 4; ++j) {
      const int n = n0 + wn + j * 16 + l15;
      const int mb = m0 + wm + i * 16 + mh;
#pragma unroll
      for (int r = 0; r < 8; ++r)
        C[(size_t)(mb + r) * N + n] = acc[i][j][r];
    }
  }
}

// ---------------------------------------------------------------------------
// RMS norm in place over q (cols 0..1023) and k (cols 1024..2047) of qkv
// ---------------------------------------------------------------------------
__global__ __launch_bounds__(256) void k_rms(
    float* __restrict__ qkv, const float* __restrict__ qw,
    const float* __restrict__ kw) {
  __shared__ float red[8];
  const int row = blockIdx.x >> 1;
  const int which = blockIdx.x & 1;
  float* p = qkv + (size_t)row * NQKV + which * DIMc;
  const float* wgt = which ? kw : qw;
  const int tid = threadIdx.x;

  float vals[4];
  float ss = 0.f;
#pragma unroll
  for (int c = 0; c < 4; ++c) {
    vals[c] = p[tid + 256 * c];
    ss += vals[c] * vals[c];
  }
#pragma unroll
  for (int off = 16; off > 0; off >>= 1) ss += __shfl_xor(ss, off, 32);
  if ((tid & 31) == 0) red[tid >> 5] = ss;
  __syncthreads();
  float total = red[0] + red[1] + red[2] + red[3] +
                red[4] + red[5] + red[6] + red[7];
  const float sc = rsqrtf(total * (1.0f / DIMc) + 1e-6f);
#pragma unroll
  for (int c = 0; c < 4; ++c)
    p[tid + 256 * c] = vals[c] * sc * wgt[tid + 256 * c];
}

// ---------------------------------------------------------------------------
// Sliding-window ALiBi attention. 32 queries/block, K/V rows [q0-16, q0+31]
// staged in LDS via async-to-LDS (48 rows x 128 f32 x2 = 48KB). One wave per
// query (4 each). Lane o holds key offset o (j = t-16+o); wave32 butterflies
// for softmax. Out-of-range rows clamp to t=0: they feed only lanes whose
// softmax weight is exactly 0 and the clamped data is finite. Writes y as
// bf16 in [B,T,H*D] layout for the output WMMA GEMM.
// ---------------------------------------------------------------------------
__global__ __launch_bounds__(256) void k_attn(
    const float* __restrict__ qkv, __bf16* __restrict__ y) {
  constexpr int TQ = 32, KR = TQ + WIN;  // 48 key rows
  __shared__ __align__(16) float Ks[KR][HDIM];
  __shared__ __align__(16) float Vs[KR][HDIM];

  const int bh = blockIdx.y;
  const int b = bh >> 3, h = bh & 7;
  const int q0 = blockIdx.x * TQ;
  const int tid = threadIdx.x;
  const int w = tid >> 5, lane = tid & 31;

  // stage: KR*HDIM/4 = 1536 16B-quads per matrix, 256 threads -> 6 iters
#pragma unroll
  for (int itr = 0; itr < (KR * HDIM / 4) / 256; ++itr) {
    const int q4 = tid + itr * 256;
    const int r = q4 >> 5;               // 32 quads per 128-float row
    const int d = (q4 & 31) * 4;
    int t = q0 - WIN + r;
    if (t < 0) t = 0;
    const float* gk =
        qkv + ((size_t)(b * Tc + t)) * NQKV + DIMc + h * HDIM + d;
    async_b128(lds_off(&Ks[r][d]), gk);
    async_b128(lds_off(&Vs[r][d]), gk + DIMc);
  }
  wait_async();
  __syncthreads();

  const float slope = exp2f(-(float)h);            // slopes[i] = 0.5^i
  const float scale = 0.088388347648318447f;       // 1/sqrt(128)

  for (int qq = 0; qq < 4; ++qq) {
    const int qi = w * 4 + qq;
    const int t = q0 + qi;
    const size_t qbase = ((size_t)(b * Tc + t)) * NQKV + h * HDIM;
    float qv[4];
#pragma unroll
    for (int c = 0; c < 4; ++c) qv[c] = qkv[qbase + lane + 32 * c];

    // scores: key offset o -> LDS row qi+o
    float myscore = -1e30f;
    for (int o = 0; o <= WIN; ++o) {
      const int r = qi + o;
      float part = 0.f;
#pragma unroll
      for (int c = 0; c < 4; ++c) part += qv[c] * Ks[r][lane + 32 * c];
#pragma unroll
      for (int off = 16; off > 0; off >>= 1) part += __shfl_xor(part, off, 32);
      if (lane == o) {
        const int j = t - WIN + o;
        myscore = (j >= 0) ? (part * scale + slope * (float)(o - WIN)) : -1e30f;
      }
    }
    const bool valid = (lane <= WIN) && ((t - WIN + lane) >= 0);
    float m = myscore;
#pragma unroll
    for (int off = 16; off > 0; off >>= 1)
      m = fmaxf(m, __shfl_xor(m, off, 32));
    const float e = valid ? __expf(myscore - m) : 0.f;
    float sum = e;
#pragma unroll
    for (int off = 16; off > 0; off >>= 1) sum += __shfl_xor(sum, off, 32);
    const float p = e / sum;

    float acc[4] = {0.f, 0.f, 0.f, 0.f};
    for (int o = 0; o <= WIN; ++o) {
      const float po = __shfl(p, o, 32);
      const int r = qi + o;
#pragma unroll
      for (int c = 0; c < 4; ++c) acc[c] += po * Vs[r][lane + 32 * c];
    }
    const size_t ybase = ((size_t)(b * Tc + t)) * DIMc + h * HDIM;
#pragma unroll
    for (int c = 0; c < 4; ++c) y[ybase + lane + 32 * c] = (__bf16)acc[c];
  }
}

// ---------------------------------------------------------------------------
// Launch
// ---------------------------------------------------------------------------
extern "C" void kernel_launch(void* const* d_in, const int* in_sizes, int n_in,
                              void* d_out, int out_size, void* d_ws,
                              size_t ws_size, hipStream_t stream) {
  const float* x  = (const float*)d_in[0];
  const float* wq = (const float*)d_in[1];
  const float* wk = (const float*)d_in[2];
  const float* wv = (const float*)d_in[3];
  const float* wo = (const float*)d_in[4];
  const float* qn = (const float*)d_in[5];
  const float* kn = (const float*)d_in[6];

  char* ws = (char*)d_ws;
  const size_t OFF_WQKV = (size_t)ROWS * NQKV * sizeof(float);        // 96MB
  const size_t OFF_WO   = OFF_WQKV + (size_t)NQKV * DIMc * 2;         // +6MB
  const size_t OFF_Y    = OFF_WO + (size_t)DIMc * DIMc * 2;           // +2MB
  const size_t OFF_XB   = OFF_Y + (size_t)ROWS * DIMc * 2;            // +16MB

  float*  qkv    = (float*)ws;
  __bf16* wqkv_t = (__bf16*)(ws + OFF_WQKV);
  __bf16* wo_t   = (__bf16*)(ws + OFF_WO);
  __bf16* ybuf   = (__bf16*)(ws + OFF_Y);
  __bf16* xbf    = (__bf16*)(ws + OFF_XB);

  k_conv_x<<<(ROWS * DIMc / 4) / 256, 256, 0, stream>>>(x, xbf);
  k_conv_wqkv<<<(NQKV * DIMc) / 256, 256, 0, stream>>>(wq, wk, wv, wqkv_t);
  k_conv_wo<<<(DIMc * DIMc) / 256, 256, 0, stream>>>(wo, wo_t);

  // x[8192,1024] @ Wqkv -> qkv[8192,3072]
  k_gemm_wmma<<<dim3(NQKV / 128, ROWS / 128), 256, 0, stream>>>(
      xbf, wqkv_t, qkv, ROWS, NQKV, DIMc);

  // RMS norm q and k halves in place
  k_rms<<<ROWS * 2, 256, 0, stream>>>(qkv, qn, kn);

  // windowed attention -> y bf16 [8192,1024]
  k_attn<<<dim3(Tc / 32, Bc * NHEADS), 256, 0, stream>>>(qkv, ybuf);

  // y @ Wo -> out[8192,1024] f32
  k_gemm_wmma<<<dim3(DIMc / 128, ROWS / 128), 256, 0, stream>>>(
      ybuf, wo_t, (float*)d_out, ROWS, DIMc, DIMc);
}
